// GCN_31413390803459
// MI455X (gfx1250) — matrix operand order repaired and checked by reading, not compiled, not run on hardware
//
#include <hip/hip_runtime.h>
#include <math.h>

typedef __attribute__((ext_vector_type(16))) _Float16 v16h;
typedef __attribute__((ext_vector_type(8)))  float    v8f;

#define NFEAT  512
#define NHID   128
#define NCLASS 40
#define BN_EPS 1e-5f

// ---------------------------------------------------------------- utilities
__global__ void zero_f32(float* __restrict__ p, long n) {
    long stride = (long)gridDim.x * blockDim.x;
    for (long i = (long)blockIdx.x * blockDim.x + threadIdx.x; i < n; i += stride)
        p[i] = 0.f;
}

__global__ void zero_f32_v4(float4* __restrict__ p, long n4) {
    long stride = (long)gridDim.x * blockDim.x;
    for (long i = (long)blockIdx.x * blockDim.x + threadIdx.x; i < n4; i += stride)
        p[i] = make_float4(0.f, 0.f, 0.f, 0.f);
}

// Pack W[K x Nreal] (row-major f32) into the per-lane WMMA B-operand layout:
// for tile nt, k-step ks: lane l holds 16 f16 values, element i -> B[k][col]
// with k = ks*32 + (l<16 ? 0 : 16) + i, col = nt*16 + (l&15). Cols >= Nreal -> 0.
__global__ void pack_b_kernel(const float* __restrict__ W, _Float16* __restrict__ Bp,
                              int K, int Nreal, int ntiles) {
    int ksteps = K >> 5;
    long total = (long)ntiles * ksteps * 512;   // halfs
    long stride = (long)gridDim.x * blockDim.x;
    for (long idx = (long)blockIdx.x * blockDim.x + threadIdx.x; idx < total; idx += stride) {
        int i    = (int)(idx & 15);
        int lane = (int)((idx >> 4) & 31);
        long blk = idx >> 9;
        int ks = (int)(blk % ksteps);
        int nt = (int)(blk / ksteps);
        int k   = (ks << 5) + ((lane < 16) ? 0 : 16) + i;
        int col = (nt << 4) + (lane & 15);
        float v = (col < Nreal) ? W[(long)k * Nreal + col] : 0.f;
        Bp[idx] = (_Float16)v;
    }
}

// ------------------------------------------------------------- WMMA GEMM
// C[M x Nreal] = A[M x K](f32) @ B (pre-packed f16) [+ bias], f32 accumulate.
// One wave computes one 16x16 output tile via v_wmma_f32_16x16x32_f16.
// A-row index is clamped (not branched): a padding lane loads a duplicate valid
// row whose products land only in D rows >= M, which are never stored. This
// keeps the inner loop branch-free (EXEC stays all-1s, no operand zero-init).
__global__ void wmma_gemm_f16(const float* __restrict__ A, const _Float16* __restrict__ Bp,
                              const float* __restrict__ bias, float* __restrict__ C,
                              int M, int K, int Nreal, int ntiles) {
    const int lane = threadIdx.x & 31;
    const int wib  = threadIdx.x >> 5;
    const int mtiles = (M + 15) >> 4;
    const int wid = blockIdx.x * (blockDim.x >> 5) + wib;
    if (wid >= mtiles * ntiles) return;
    const int mt = wid / ntiles;
    const int nt = wid - mt * ntiles;

    int arow_idx = (mt << 4) + (lane & 15);
    if (arow_idx >= M) arow_idx = M - 1;           // clamp, never stored
    const int kA     = (lane < 16) ? 0 : 16;       // A-operand K sub-base per lane group
    const int ksteps = K >> 5;

    const float* __restrict__ arow  = A + (long)arow_idx * K;
    const v16h*  __restrict__ bbase = (const v16h*)(Bp + (long)nt * ksteps * 512);

    v8f c = {};
    for (int ks = 0; ks < ksteps; ++ks) {
        const float* ap = arow + (ks << 5) + kA;
        v16h a;
        #pragma unroll
        for (int i = 0; i < 8; ++i) a[i]     = (_Float16)ap[i];
        #pragma unroll
        for (int i = 0; i < 8; ++i) a[8 + i] = (_Float16)ap[8 + i];
        v16h b = bbase[ks * 32 + lane];
        c = __builtin_amdgcn_wmma_f32_16x16x32_f16(false, a, false, b,
                                                   (short)0, c, false, false);
    }

    const int col = (nt << 4) + (lane & 15);
    const int r0  = (lane < 16) ? 0 : 8;
    if (col < Nreal) {
        const float bv = bias ? bias[col] : 0.f;
        #pragma unroll
        for (int r = 0; r < 8; ++r) {
            int m = (mt << 4) + r0 + r;
            if (m < M) C[(long)m * Nreal + col] = c[r] + bv;
        }
    }
}

// ---------------------------------------------------------- graph kernels
__global__ void degree_kernel(const int* __restrict__ col, const float* __restrict__ ew,
                              float* __restrict__ deg, int E) {
    int stride = gridDim.x * blockDim.x;
    for (int e = blockIdx.x * blockDim.x + threadIdx.x; e < E; e += stride)
        atomicAdd(&deg[col[e]], ew[e]);
}

__global__ void dinv_kernel(float* __restrict__ deg, int N) {
    int stride = gridDim.x * blockDim.x;
    for (int i = blockIdx.x * blockDim.x + threadIdx.x; i < N; i += stride)
        deg[i] = rsqrtf(deg[i] + 1.0f);   // +1 = self-loop weight; always > 0
}

// agg[col] += h[row] * (dinv[row]*ew*dinv[col]) ; one wave per edge, 128 channels.
// Each lane moves a float4 (channels lane*4..lane*4+3): one b128 load + 4 atomics.
__global__ void aggregate_kernel(const int* __restrict__ rows, const int* __restrict__ cols,
                                 const float* __restrict__ ew, const float* __restrict__ dinv,
                                 const float* __restrict__ h, float* __restrict__ agg, int E) {
    const int lane = threadIdx.x & 31;
    const int wpb  = blockDim.x >> 5;
    const int wstride = gridDim.x * wpb;
    for (int e0 = blockIdx.x * wpb + (threadIdx.x >> 5); e0 < E; e0 += wstride) {
        const int e = __builtin_amdgcn_readfirstlane(e0);   // wave-scalar edge id
        const int r = rows[e], c = cols[e];
        const float nrm = dinv[r] * ew[e] * dinv[c];
        const float4 hv = *(const float4*)(h + (long)r * NHID + lane * 4);
        float* __restrict__ ac = agg + (long)c * NHID + lane * 4;
        atomicAdd(ac + 0, hv.x * nrm);
        atomicAdd(ac + 1, hv.y * nrm);
        atomicAdd(ac + 2, hv.z * nrm);
        atomicAdd(ac + 3, hv.w * nrm);
    }
}

// y += self-loop msg + bias ; optional relu (in place on agg buffer), float4-wide
__global__ void finalize_conv(const float4* __restrict__ h, const float* __restrict__ dinv,
                              const float4* __restrict__ bias4, float4* __restrict__ y,
                              int N, int relu) {
    long total = (long)N * (NHID / 4);
    long stride = (long)gridDim.x * blockDim.x;
    for (long idx = (long)blockIdx.x * blockDim.x + threadIdx.x; idx < total; idx += stride) {
        int node = (int)(idx >> 5);                // NHID/4 == 32 vec4 per node
        int c4   = (int)(idx & 31);
        float di = dinv[node];
        float s  = di * di;
        float4 hv = h[idx], b = bias4[c4], v = y[idx];
        v.x += hv.x * s + b.x;  v.y += hv.y * s + b.y;
        v.z += hv.z * s + b.z;  v.w += hv.w * s + b.w;
        if (relu) {
            v.x = fmaxf(v.x, 0.f); v.y = fmaxf(v.y, 0.f);
            v.z = fmaxf(v.z, 0.f); v.w = fmaxf(v.w, 0.f);
        }
        y[idx] = v;
    }
}

// ---------------------------------------------------------------- batchnorm
__global__ void bn_stats(const float* __restrict__ y, float* __restrict__ sum,
                         float* __restrict__ sumsq, int N) {
    int c = threadIdx.x;                 // 128 threads = channels
    float s = 0.f, q = 0.f;
    for (int i = blockIdx.x; i < N; i += gridDim.x) {
        float v = y[(long)i * NHID + c];
        s += v; q += v * v;
    }
    atomicAdd(&sum[c], s);
    atomicAdd(&sumsq[c], q);
}

__global__ void bn_final(const float* __restrict__ sum, const float* __restrict__ sumsq,
                         const float* __restrict__ g, const float* __restrict__ be,
                         float* __restrict__ scale, float* __restrict__ shift, int N) {
    int c = threadIdx.x;
    float inv_n = 1.f / (float)N;
    float mu  = sum[c] * inv_n;
    float var = sumsq[c] * inv_n - mu * mu;   // biased variance
    float sc  = g[c] * rsqrtf(var + BN_EPS);
    scale[c] = sc;
    shift[c] = be[c] - mu * sc;
}

__global__ void bn_apply(const float4* __restrict__ y, const float4* __restrict__ scale4,
                         const float4* __restrict__ shift4, float4* __restrict__ out, int N) {
    long total = (long)N * (NHID / 4);
    long stride = (long)gridDim.x * blockDim.x;
    for (long idx = (long)blockIdx.x * blockDim.x + threadIdx.x; idx < total; idx += stride) {
        int c4 = (int)(idx & 31);
        float4 v = y[idx], sc = scale4[c4], sh = shift4[c4];
        v.x = v.x * sc.x + sh.x;  v.y = v.y * sc.y + sh.y;
        v.z = v.z * sc.z + sh.z;  v.w = v.w * sc.w + sh.w;
        out[idx] = v;
    }
}

// ---------------------------------------------------------------- softmax
__global__ void softmax40_kernel(const float* __restrict__ logits,
                                 float* __restrict__ probs, int N) {
    const int lane = threadIdx.x & 31;
    const int wpb  = blockDim.x >> 5;
    const int wstride = gridDim.x * wpb;
    for (int node = blockIdx.x * wpb + (threadIdx.x >> 5); node < N; node += wstride) {
        const float* L = logits + (long)node * NCLASS;
        float a = L[lane];                                         // classes 0..31
        float b = (lane < NCLASS - 32) ? L[lane + 32] : -3.4e38f;  // classes 32..39
        float m = fmaxf(a, b);
        #pragma unroll
        for (int off = 16; off; off >>= 1) m = fmaxf(m, __shfl_xor(m, off, 32));
        float ea = __expf(a - m);
        float eb = (lane < NCLASS - 32) ? __expf(b - m) : 0.f;
        float s = ea + eb;
        #pragma unroll
        for (int off = 16; off; off >>= 1) s += __shfl_xor(s, off, 32);
        float inv = 1.f / s;
        probs[(long)node * NCLASS + lane] = ea * inv;
        if (lane < NCLASS - 32) probs[(long)node * NCLASS + lane + 32] = eb * inv;
    }
}

// ================================================================= launch
extern "C" void kernel_launch(void* const* d_in, const int* in_sizes, int n_in,
                              void* d_out, int out_size, void* d_ws, size_t ws_size,
                              hipStream_t stream) {
    const float* x    = (const float*)d_in[0];
    const int*   ei   = (const int*)d_in[1];
    const float* ew   = (const float*)d_in[2];
    const float* W1   = (const float*)d_in[3];
    const float* b1   = (const float*)d_in[4];
    const float* W2   = (const float*)d_in[5];
    const float* b2   = (const float*)d_in[6];
    const float* W3   = (const float*)d_in[7];
    const float* b3   = (const float*)d_in[8];
    const float* g1   = (const float*)d_in[9];
    const float* be1  = (const float*)d_in[10];
    const float* g2   = (const float*)d_in[11];
    const float* be2  = (const float*)d_in[12];
    const float* g3   = (const float*)d_in[13];
    const float* be3  = (const float*)d_in[14];
    const float* linW = (const float*)d_in[15];
    const float* linb = (const float*)d_in[16];

    const int N = in_sizes[0] / NFEAT;   // 50000
    const int E = in_sizes[2];           // 800000
    const int* rows = ei;                // edge_index[0]
    const int* cols = ei + E;            // edge_index[1]

    // --------- workspace carve (256B aligned) ----------
    char* wsp = (char*)d_ws;
    auto alloc = [&](size_t bytes) -> void* {
        void* p = wsp; wsp += (bytes + 255) & ~(size_t)255; return p;
    };
    float*    dinv = (float*)alloc((size_t)N * 4);
    float*    h    = (float*)alloc((size_t)N * NHID * 4);   // GEMM output
    float*    y    = (float*)alloc((size_t)N * NHID * 4);   // agg / conv output
    float*    xb   = (float*)alloc((size_t)N * NHID * 4);   // BN output (layer input)
    _Float16* Bp1  = (_Float16*)alloc((size_t)8 * (NFEAT / 32) * 512 * 2);
    _Float16* Bp2  = (_Float16*)alloc((size_t)8 * (NHID / 32) * 512 * 2);
    _Float16* Bp3  = (_Float16*)alloc((size_t)8 * (NHID / 32) * 512 * 2);
    _Float16* BpL  = (_Float16*)alloc((size_t)3 * (NHID / 32) * 512 * 2);
    float*    sums = (float*)alloc(256 * 4);   // sum[128] | sumsq[128]
    float*    sca  = (float*)alloc(256 * 4);   // scale[128] | shift[128]

    float* logits = (float*)d_out;
    float* probs  = logits + (long)N * NCLASS;
    float* x3     = probs  + (long)N * NCLASS;

    // --------- weight packing (once per call, cheap) ----------
    pack_b_kernel<<<256, 256, 0, stream>>>(W1,   Bp1, NFEAT, NHID,   8);
    pack_b_kernel<<<64,  256, 0, stream>>>(W2,   Bp2, NHID,  NHID,   8);
    pack_b_kernel<<<64,  256, 0, stream>>>(W3,   Bp3, NHID,  NHID,   8);
    pack_b_kernel<<<32,  256, 0, stream>>>(linW, BpL, NHID,  NCLASS, 3);

    // --------- gcn_norm ----------
    zero_f32<<<256, 256, 0, stream>>>(dinv, N);
    degree_kernel<<<1024, 256, 0, stream>>>(cols, ew, dinv, E);
    dinv_kernel<<<256, 256, 0, stream>>>(dinv, N);

    const int mtiles = (N + 15) / 16;

    auto run_layer = [&](const float* xin, int K, const _Float16* Bp,
                         const float* bias, const float* g, const float* be,
                         int relu, float* out) {
        int tiles = mtiles * 8;
        wmma_gemm_f16<<<(tiles + 7) / 8, 256, 0, stream>>>(xin, Bp, nullptr, h,
                                                           N, K, NHID, 8);
        zero_f32_v4<<<2048, 256, 0, stream>>>((float4*)y, (long)N * (NHID / 4));
        aggregate_kernel<<<(E + 7) / 8, 256, 0, stream>>>(rows, cols, ew, dinv, h, y, E);
        finalize_conv<<<2048, 256, 0, stream>>>((const float4*)h, dinv,
                                                (const float4*)bias, (float4*)y, N, relu);
        zero_f32<<<1, 256, 0, stream>>>(sums, 256);
        bn_stats<<<512, 128, 0, stream>>>(y, sums, sums + 128, N);
        bn_final<<<1, 128, 0, stream>>>(sums, sums + 128, g, be, sca, sca + 128, N);
        bn_apply<<<2048, 256, 0, stream>>>((const float4*)y, (const float4*)sca,
                                           (const float4*)(sca + 128), (float4*)out, N);
    };

    run_layer(x,  NFEAT, Bp1, b1, g1, be1, 1, xb);
    run_layer(xb, NHID,  Bp2, b2, g2, be2, 1, xb);
    run_layer(xb, NHID,  Bp3, b3, g3, be3, 0, x3);

    // --------- final linear (N padded to 48, col<40 predicated) + softmax ----------
    int tilesL = mtiles * 3;
    wmma_gemm_f16<<<(tilesL + 7) / 8, 256, 0, stream>>>(x3, BpL, linb, logits,
                                                        N, NHID, NCLASS, 3);
    softmax40_kernel<<<(N + 7) / 8, 256, 0, stream>>>(logits, probs, N);
}
